// CausalCrossConditionalSelfAttention_83047487635486
// MI455X (gfx1250) — matrix operand order repaired and checked by reading, not compiled, not run on hardware
//
#include <hip/hip_runtime.h>

typedef __attribute__((ext_vector_type(16))) _Float16 v16h;
typedef __attribute__((ext_vector_type(8)))  float    v8f;

// Problem constants fixed by setup_inputs()
constexpr int B_  = 2;
constexpr int T_  = 1024;
constexpr int N_  = 16;
constexpr int C_  = 512;
constexpr int H_  = 8;
constexpr int HD  = 64;                // C_/H_
constexpr int L_  = 3 * T_ + 4 * N_;   // 3136
constexpr int LT  = L_ / 16;           // 196 row tiles (3T boundary at tile 192)

// ---------------------------------------------------------------------------
// CDNA5 async global->LDS copy (ASYNCcnt-tracked).  lds = low 32 bits of the
// generic pointer (ISA 10.2: flat->LDS uses addr[31:0]); 16 bytes per lane.
// ---------------------------------------------------------------------------
__device__ __forceinline__ void async_b128(uint32_t lds_byte, const void* g)
{
    asm volatile("global_load_async_to_lds_b128 %0, %1, off"
                 :: "v"(lds_byte), "v"((unsigned long long)(uintptr_t)g)
                 : "memory");
}
__device__ __forceinline__ void wait_async0()
{
    asm volatile("s_wait_asynccnt 0" ::: "memory");
}
__device__ __forceinline__ uint32_t lds_addr(const void* p)
{
    return (uint32_t)(uintptr_t)p;     // low 32 bits = LDS byte offset
}

// ---------------------------------------------------------------------------
// Kernel 1: fused QKV projection.  out = (x @ W^T + b) [ * 1/8 for Q ]
// One wave computes a 16x64 tile (one head's columns): 4 wmma per K-step,
// A-tile reused.  Results stored f16 as [B][H][L][HD].
// ---------------------------------------------------------------------------
__global__ __launch_bounds__(32) void qkv_kernel(
    const float* __restrict__ x,
    const float* __restrict__ Wq, const float* __restrict__ bq,
    const float* __restrict__ Wk, const float* __restrict__ bk,
    const float* __restrict__ Wv, const float* __restrict__ bv,
    _Float16* __restrict__ qbuf, _Float16* __restrict__ kbuf,
    _Float16* __restrict__ vbuf)
{
    const int lt  = blockIdx.x;           // 0..195 row tile
    const int ct  = blockIdx.y;           // 0..7   head / 64-col tile
    const int zb  = blockIdx.z;           // mat*B + b
    const int mat = zb / B_;
    const int b   = zb - mat * B_;

    const float* W    = (mat == 0) ? Wq : (mat == 1) ? Wk : Wv;
    const float* bias = (mat == 0) ? bq : (mat == 1) ? bk : bv;
    _Float16*    obuf = (mat == 0) ? qbuf : (mat == 1) ? kbuf : vbuf;
    const float  scl  = (mat == 0) ? 0.125f : 1.0f;   // fold 1/sqrt(hd) into Q

    __shared__ _Float16 la[16][40];   // A tile 16x32 (row = 80B, 16B aligned)
    __shared__ _Float16 lb[32][68];   // B tile 32x64 (padded)

    const int lane = threadIdx.x;
    const int m    = lane & 15;
    const int lh   = lane >> 4;
    const int l0   = lt * 16, c0 = ct * 64;

    v8f acc[4] = {};
    for (int k0 = 0; k0 < C_; k0 += 32) {
        // stage A = x[l0:l0+16, k0:k0+32] (f32->f16), 128 x float4 (4/lane)
#pragma unroll
        for (int u = 0; u < 4; ++u) {
            int i = lane + u * 32;
            int r = i >> 3, c4 = (i & 7) << 2;
            const float4 f =
                *(const float4*)&x[(size_t)(b * L_ + l0 + r) * C_ + k0 + c4];
            la[r][c4]     = (_Float16)f.x;
            la[r][c4 + 1] = (_Float16)f.y;
            la[r][c4 + 2] = (_Float16)f.z;
            la[r][c4 + 3] = (_Float16)f.w;
        }
        // stage B = W^T[k0:k0+32, c0:c0+64]:  lb[kk][n] = W[c0+n][k0+kk]
#pragma unroll
        for (int u = 0; u < 16; ++u) {
            int i = lane + u * 32;
            int n = i >> 3, c4 = (i & 7) << 2;
            const float4 f =
                *(const float4*)&W[(size_t)(c0 + n) * C_ + k0 + c4];
            lb[c4][n]     = (_Float16)f.x;
            lb[c4 + 1][n] = (_Float16)f.y;
            lb[c4 + 2][n] = (_Float16)f.z;
            lb[c4 + 3][n] = (_Float16)f.w;
        }
        __syncthreads();

        v16h a;
#pragma unroll
        for (int v = 0; v < 8; ++v) {
            // A 16x32 f16 layout: K = grp*16 + half*8 + sub*2
            int ka = (v >> 2) * 16 + lh * 8 + (v & 3) * 2;
            a[2 * v]     = la[m][ka];
            a[2 * v + 1] = la[m][ka + 1];
        }
#pragma unroll
        for (int j = 0; j < 4; ++j) {
            v16h bm;
#pragma unroll
            for (int v = 0; v < 8; ++v) {
                int kb = (v >> 2) * 16 + lh * 8 + (v & 3) * 2;  // B: K = grp*16 + half*8 + sub*2
                bm[2 * v]     = lb[kb][j * 16 + m];
                bm[2 * v + 1] = lb[kb + 1][j * 16 + m];
            }
            acc[j] = __builtin_amdgcn_wmma_f32_16x16x32_f16(
                false, a, false, bm, (short)0, acc[j], false, false);
        }
        __syncthreads();
    }

    // C/D layout: row = r + 8*half, col = j*16 + (lane&15); head = ct
    _Float16* og = obuf + (((size_t)b * H_ + ct) * L_) * HD;
#pragma unroll
    for (int j = 0; j < 4; ++j) {
        int d = j * 16 + m;
        float bb = bias[c0 + d];
#pragma unroll
        for (int r = 0; r < 8; ++r) {
            int row = l0 + r + 8 * lh;
            og[(size_t)row * HD + d] = (_Float16)((acc[j][r] + bb) * scl);
        }
    }
}

// ---------------------------------------------------------------------------
// Structured mask (True = masked).  br = query block (0,1,2), fr = frame row.
// ---------------------------------------------------------------------------
__device__ __forceinline__ bool is_masked(int br, int fr, int c,
                                          const int* rs, const int* re)
{
    if (c < 3 * T_) {
        int bc = c / T_;
        int fc = c - bc * T_;
        bool un;
        if (br == 0)      un = (bc == 0) && (fc <= fr);
        else if (br == 1) un = (bc == 0) ? (fc <= fr) : (fc < fr);
        else              un = (bc <= 1) ? (fc <= fr) : (fc < fr);
        return !un;
    } else {
        if (br == 0) return true;                   // upper-motion rows never see text
        int j   = c - 3 * T_;
        int seg = j >> 4, n = j & 15;               // upper|lower|torso|whole
        bool act   = (fr >= rs[n]) && (fr < re[n]);
        bool segok = (br == 1) ? (seg != 1) : (seg != 0);
        return !(act && segok);
    }
}

// ---------------------------------------------------------------------------
// Kernel 2: flash attention.  One wave = 16 queries for one (b,h).
// Q/K/V tiles are contiguous 2-4KB blocks in [B,H,L,64] f16 -> staged with
// global_load_async_to_lds_b128 (ASYNCcnt).  Key loop in chunks of 32:
// S = Q(16x64)Kt (4 wmma), mask + online softmax, O += P(16x32)V (4 wmma).
// ---------------------------------------------------------------------------
__global__ __launch_bounds__(32) void attn_kernel(
    const _Float16* __restrict__ qbuf, const _Float16* __restrict__ kbuf,
    const _Float16* __restrict__ vbuf, _Float16* __restrict__ ybuf,
    const int* __restrict__ sfr, const int* __restrict__ efr,
    const int* __restrict__ w0p)
{
    const int qt = blockIdx.x;   // 0..195
    const int h  = blockIdx.y;
    const int b  = blockIdx.z;
    const int lane = threadIdx.x;
    const int m  = lane & 15;
    const int lh = lane >> 4;

    __shared__ _Float16 qld[16][64];   // unpadded: async-copied 1:1
    __shared__ _Float16 kt[32][64];
    __shared__ _Float16 vt[32][64];
    __shared__ _Float16 pt[16][40];
    __shared__ int rs[16], re[16];

    const int w0 = w0p[0];
    if (lane < 16) {
        rs[lane] = sfr[b * N_ + lane] / 8 - w0;
        re[lane] = efr[b * N_ + lane] / 8 - w0;
    }

    const size_t base = ((size_t)b * H_ + h) * L_;
    const _Float16* Qg = qbuf + base * HD;
    const _Float16* Kg = kbuf + base * HD;
    const _Float16* Vg = vbuf + base * HD;

    const uint32_t qoff = lds_addr(&qld[0][0]);
    const uint32_t koff = lds_addr(&kt[0][0]);
    const uint32_t voff = lds_addr(&vt[0][0]);

    const int q0 = qt * 16;
    // stage Q tile: 2KB contiguous (4 x b128 per lane)
#pragma unroll
    for (int u = 0; u < 4; ++u) {
        int i = lane + u * 32;
        async_b128(qoff + i * 16, Qg + (size_t)q0 * HD + i * 8);
    }
    wait_async0();
    __syncthreads();

    v16h qa[2];
#pragma unroll
    for (int t = 0; t < 2; ++t)
#pragma unroll
        for (int v = 0; v < 8; ++v) {
            int kk = t * 32 + (v >> 2) * 16 + lh * 8 + (v & 3) * 2;
            qa[t][2 * v]     = qld[m][kk];
            qa[t][2 * v + 1] = qld[m][kk + 1];
        }

    v8f  o[4] = {};
    float mi[8], li[8];
#pragma unroll
    for (int r = 0; r < 8; ++r) { mi[r] = -1e30f; li[r] = 0.0f; }

    const bool qIsText = (q0 >= 3 * T_);      // tiles never straddle 3T (192*16)
    const int  br  = qIsText ? 0 : (q0 / T_);
    const int  fr0 = qIsText ? 0 : (q0 - br * T_);

    for (int kb = 0; kb < L_; kb += 32) {
        __syncthreads();
        // stage K,V chunks: two contiguous 4KB blocks (8+8 b128 per lane)
#pragma unroll
        for (int u = 0; u < 8; ++u) {
            int i = lane + u * 32;
            async_b128(koff + i * 16, Kg + (size_t)kb * HD + i * 8);
            async_b128(voff + i * 16, Vg + (size_t)kb * HD + i * 8);
        }
        wait_async0();
        __syncthreads();

        // ---- S = Q . K^T  (two 16-key groups, K-dim 64 split into 2x32) ----
        v8f s01[2];
#pragma unroll
        for (int g = 0; g < 2; ++g) {
            v8f sg = {};
#pragma unroll
            for (int t = 0; t < 2; ++t) {
                v16h bk;
#pragma unroll
                for (int v = 0; v < 8; ++v) {
                    int kk = t * 32 + (v >> 2) * 16 + lh * 8 + (v & 3) * 2;   // d index
                    bk[2 * v]     = kt[g * 16 + m][kk];  // B[d][key]=K[key][d]
                    bk[2 * v + 1] = kt[g * 16 + m][kk + 1];
                }
                sg = __builtin_amdgcn_wmma_f32_16x16x32_f16(
                    false, qa[t], false, bk, (short)0, sg, false, false);
            }
            s01[g] = sg;
        }

        // ---- mask + online softmax (rows live in 16-lane halves) ----
        float pv0[8], pv1[8], alpha[8];
#pragma unroll
        for (int r = 0; r < 8; ++r) {
            float v0 = s01[0][r], v1 = s01[1][r];
            if (!qIsText) {
                int fr = fr0 + r + 8 * lh;
                if (is_masked(br, fr, kb + m,      rs, re)) v0 = -1e30f;
                if (is_masked(br, fr, kb + 16 + m, rs, re)) v1 = -1e30f;
            }
            float t = fmaxf(v0, v1);
            t = fmaxf(t, __shfl_xor(t, 1, 32));
            t = fmaxf(t, __shfl_xor(t, 2, 32));
            t = fmaxf(t, __shfl_xor(t, 4, 32));
            t = fmaxf(t, __shfl_xor(t, 8, 32));
            float mn = fmaxf(mi[r], t);
            float al = __expf(mi[r] - mn);
            float p0 = (v0 < -1e29f) ? 0.0f : __expf(v0 - mn);
            float p1 = (v1 < -1e29f) ? 0.0f : __expf(v1 - mn);
            float sm = p0 + p1;
            sm += __shfl_xor(sm, 1, 32);
            sm += __shfl_xor(sm, 2, 32);
            sm += __shfl_xor(sm, 4, 32);
            sm += __shfl_xor(sm, 8, 32);
            li[r] = li[r] * al + sm;
            mi[r] = mn;
            alpha[r] = al; pv0[r] = p0; pv1[r] = p1;
        }

        // rescale O, publish P in C/D layout -> LDS
#pragma unroll
        for (int r = 0; r < 8; ++r) {
#pragma unroll
            for (int j = 0; j < 4; ++j) o[j][r] *= alpha[r];
            int mrow = r + 8 * lh;
            pt[mrow][m]      = (_Float16)pv0[r];
            pt[mrow][16 + m] = (_Float16)pv1[r];
        }
        __syncthreads();

        // reload P in A-matrix layout
        v16h pa;
#pragma unroll
        for (int v = 0; v < 8; ++v) {
            int kk = (v >> 2) * 16 + lh * 8 + (v & 3) * 2;
            pa[2 * v]     = pt[m][kk];
            pa[2 * v + 1] = pt[m][kk + 1];
        }

        // ---- O += P . V  (K-dim = 32 keys, 4 column groups of hd) ----
#pragma unroll
        for (int j = 0; j < 4; ++j) {
            v16h vb;
#pragma unroll
            for (int v = 0; v < 8; ++v) {
                int kk = (v >> 2) * 16 + lh * 8 + (v & 3) * 2;            // key index within chunk
                vb[2 * v]     = vt[kk][j * 16 + m];
                vb[2 * v + 1] = vt[kk + 1][j * 16 + m];
            }
            o[j] = __builtin_amdgcn_wmma_f32_16x16x32_f16(
                false, pa, false, vb, (short)0, o[j], false, false);
        }
    }

    // normalize and store: y[b][q0+row][h*64 + j*16 + n]
    _Float16* Yg = ybuf + (size_t)b * L_ * C_ + h * HD;
#pragma unroll
    for (int r = 0; r < 8; ++r) {
        int   row = q0 + r + 8 * lh;
        float inv = 1.0f / li[r];
#pragma unroll
        for (int j = 0; j < 4; ++j)
            Yg[(size_t)row * C_ + j * 16 + m] = (_Float16)(o[j][r] * inv);
    }
}

// ---------------------------------------------------------------------------
// Kernel 3: output projection.  out = y @ Wp^T + bp  (f16 in, f32 out)
// One wave computes a 16x64 tile: 4 wmma per K-step.
// ---------------------------------------------------------------------------
__global__ __launch_bounds__(32) void proj_kernel(
    const _Float16* __restrict__ ybuf, const float* __restrict__ Wp,
    const float* __restrict__ bp, float* __restrict__ out)
{
    const int lt = blockIdx.x, ct = blockIdx.y, b = blockIdx.z;
    __shared__ _Float16 la[16][40];
    __shared__ _Float16 lb[32][68];
    const int lane = threadIdx.x;
    const int m = lane & 15, lh = lane >> 4;
    const int l0 = lt * 16, c0 = ct * 64;

    v8f acc[4] = {};
    for (int k0 = 0; k0 < C_; k0 += 32) {
        // A: f16 rows, 16B vector copies (64 x uint4, 2/lane)
#pragma unroll
        for (int u = 0; u < 2; ++u) {
            int i = lane + u * 32;
            int r = i >> 2, c8 = (i & 3) << 3;
            const uint4 v =
                *(const uint4*)&ybuf[(size_t)(b * L_ + l0 + r) * C_ + k0 + c8];
            *(uint4*)&la[r][c8] = v;
        }
        // B: Wp^T tile 32x64 (16 float4 per lane)
#pragma unroll
        for (int u = 0; u < 16; ++u) {
            int i = lane + u * 32;
            int n = i >> 3, c4 = (i & 7) << 2;
            const float4 f =
                *(const float4*)&Wp[(size_t)(c0 + n) * C_ + k0 + c4];
            lb[c4][n]     = (_Float16)f.x;
            lb[c4 + 1][n] = (_Float16)f.y;
            lb[c4 + 2][n] = (_Float16)f.z;
            lb[c4 + 3][n] = (_Float16)f.w;
        }
        __syncthreads();

        v16h a;
#pragma unroll
        for (int v = 0; v < 8; ++v) {
            int ka = (v >> 2) * 16 + lh * 8 + (v & 3) * 2;
            a[2 * v]     = la[m][ka];
            a[2 * v + 1] = la[m][ka + 1];
        }
#pragma unroll
        for (int j = 0; j < 4; ++j) {
            v16h bm;
#pragma unroll
            for (int v = 0; v < 8; ++v) {
                int kb = (v >> 2) * 16 + lh * 8 + (v & 3) * 2;
                bm[2 * v]     = lb[kb][j * 16 + m];
                bm[2 * v + 1] = lb[kb + 1][j * 16 + m];
            }
            acc[j] = __builtin_amdgcn_wmma_f32_16x16x32_f16(
                false, a, false, bm, (short)0, acc[j], false, false);
        }
        __syncthreads();
    }
#pragma unroll
    for (int j = 0; j < 4; ++j) {
        int col = c0 + j * 16 + m;
        float bb = bp[col];
#pragma unroll
        for (int r = 0; r < 8; ++r) {
            int row = l0 + r + 8 * lh;
            out[(size_t)(b * L_ + row) * C_ + col] = acc[j][r] + bb;
        }
    }
}

// ---------------------------------------------------------------------------
extern "C" void kernel_launch(void* const* d_in, const int* in_sizes, int n_in,
                              void* d_out, int out_size, void* d_ws, size_t ws_size,
                              hipStream_t stream)
{
    const float* x  = (const float*)d_in[0];
    const float* Wq = (const float*)d_in[1];
    const float* bq = (const float*)d_in[2];
    const float* Wk = (const float*)d_in[3];
    const float* bk = (const float*)d_in[4];
    const float* Wv = (const float*)d_in[5];
    const float* bv = (const float*)d_in[6];
    const float* Wp = (const float*)d_in[7];
    const float* bp = (const float*)d_in[8];
    const int*   sf = (const int*)d_in[9];
    const int*   ef = (const int*)d_in[10];
    const int*   w0 = (const int*)d_in[13];   // window_abs_start (device scalar)

    _Float16* ws = (_Float16*)d_ws;
    const size_t qkvN = (size_t)B_ * H_ * L_ * HD;   // == B*L*C elements
    _Float16* qb = ws;
    _Float16* kb = ws + qkvN;
    _Float16* vb = ws + 2 * qkvN;
    _Float16* yb = ws + 3 * qkvN;                    // ~25.7 MB f16 total

    dim3 g1(LT, H_, 3 * B_);
    qkv_kernel<<<g1, 32, 0, stream>>>(x, Wq, bq, Wk, bk, Wv, bv, qb, kb, vb);

    dim3 g2(LT, H_, B_);
    attn_kernel<<<g2, 32, 0, stream>>>(qb, kb, vb, yb, sf, ef, w0);

    dim3 g3(LT, H_, B_);
    proj_kernel<<<g3, 32, 0, stream>>>(yb, Wp, bp, (float*)d_out);
}
